// MultiHeadAttention_57698590655007
// MI455X (gfx1250) — matrix operand order repaired and checked by reading, not compiled, not run on hardware
//
#include <hip/hip_runtime.h>

// ---------------- problem constants ----------------
#define Bc  4
#define Sc  2048
#define Dc  768
#define Hc  12
#define DHc 64
#define Mrows (Bc*Sc)          // 8192

typedef __attribute__((ext_vector_type(16))) __bf16          v16bf;
typedef __attribute__((ext_vector_type(8)))  float           v8f;
typedef __attribute__((ext_vector_type(8)))  unsigned short  ushort8;
typedef __attribute__((ext_vector_type(16))) unsigned short  ushort16;

// fp32 -> bf16, round to nearest even
__device__ inline unsigned short f2bf(float f) {
  unsigned int u = __builtin_bit_cast(unsigned int, f);
  unsigned int r = u + 0x7FFFu + ((u >> 16) & 1u);
  return (unsigned short)(r >> 16);
}

__device__ inline v16bf make_frag(ushort8 lo, ushort8 hi) {
  ushort16 t;
#pragma unroll
  for (int i = 0; i < 8; ++i) { t[i] = lo[i]; t[i + 8] = hi[i]; }
  return __builtin_bit_cast(v16bf, t);
}

__device__ inline v16bf frag16(const unsigned short* p) {
  return __builtin_bit_cast(v16bf, *(const ushort16*)p);
}

__device__ inline v8f wmma_bf16(v16bf a, v16bf b, v8f c) {
  return __builtin_amdgcn_wmma_f32_16x16x32_bf16(
      false, a, false, b, (short)0, c, false, false);
}

// async DMA: 16B global -> LDS, tracked by ASYNCcnt (CDNA5 path)
__device__ inline void async_load16(unsigned lds_off, const void* g) {
  unsigned long long ga = (unsigned long long)g;
  asm volatile("global_load_async_to_lds_b128 %0, %1, off"
               :: "v"(lds_off), "v"(ga) : "memory");
}
// low 32 bits of a flat shared-aperture address == LDS byte offset (ISA aperture rule)
__device__ inline unsigned lds_addr(const void* p) {
  return (unsigned)(unsigned long long)p;
}

// ---------------- conversion kernels ----------------
__global__ void k_cvt(const float* __restrict__ in, unsigned short* __restrict__ out, int n) {
  int i = blockIdx.x * blockDim.x + threadIdx.x;
  if (i < n) out[i] = f2bf(in[i]);
}

// W [Hh, Dd, Nn] (fp32) -> WT [Hh, Nn, Dd] (bf16)
__global__ void k_cvt_wt(const float* __restrict__ W, unsigned short* __restrict__ WT,
                         int Hh, int Dd, int Nn) {
  int i = blockIdx.x * blockDim.x + threadIdx.x;
  int total = Hh * Dd * Nn;
  if (i >= total) return;
  int n = i % Nn;
  int d = (i / Nn) % Dd;
  int h = i / (Nn * Dd);
  WT[((size_t)h * Nn + n) * Dd + d] = f2bf(W[i]);
}

// ---------------- QKV projection ----------------
// grid: (Mrows/128, H, 3); block 256 (8 waves). Wave w: rows m0..m0+15, all 64 cols.
__global__ __launch_bounds__(256) void k_qkv(
    const unsigned short* __restrict__ xb,
    const unsigned short* __restrict__ WqT, const unsigned short* __restrict__ WkT,
    const unsigned short* __restrict__ WvT,
    const float* __restrict__ bq, const float* __restrict__ bk, const float* __restrict__ bvv,
    unsigned short* __restrict__ Q, unsigned short* __restrict__ K,
    unsigned short* __restrict__ VT) {
  int wave = threadIdx.x >> 5;
  int lane = threadIdx.x & 31;
  int half = lane >> 4;
  int ln   = lane & 15;
  int m0   = blockIdx.x * 128 + wave * 16;
  int h    = blockIdx.y;
  int which = blockIdx.z;   // 0=Q 1=K 2=V

  const unsigned short* WT   = which == 0 ? WqT : (which == 1 ? WkT : WvT);
  const float*          bias = which == 0 ? bq  : (which == 1 ? bk  : bvv);

  const unsigned short* arow  = xb + (size_t)(m0 + ln) * Dc;
  const unsigned short* wbase = WT + (size_t)h * DHc * Dc;

  v8f acc[4] = {};
  for (int kc = 0; kc < Dc; kc += 32) {
    v16bf a = frag16(arow + kc + 16 * half);
#pragma unroll
    for (int nt = 0; nt < 4; ++nt) {
      const unsigned short* bp = wbase + (size_t)(nt * 16 + ln) * Dc + kc + 16 * half;
      acc[nt] = wmma_bf16(a, frag16(bp), acc[nt]);
    }
  }

  int bidx = m0 / Sc;
  int srow = m0 % Sc;
  if (which < 2) {
    unsigned short* out = (which == 0) ? Q : K;
#pragma unroll
    for (int nt = 0; nt < 4; ++nt) {
      float bb = bias[h * DHc + nt * 16 + ln];
      size_t base = (((size_t)bidx * Hc + h) * Sc + srow + 8 * half) * DHc + nt * 16 + ln;
#pragma unroll
      for (int r = 0; r < 8; ++r)
        out[base + (size_t)r * DHc] = f2bf(acc[nt][r] + bb);
    }
  } else {
    // V stored transposed: VT[((b*H+h)*DH + n)*S + s]; contiguous over r (s)
#pragma unroll
    for (int nt = 0; nt < 4; ++nt) {
      float bb = bias[h * DHc + nt * 16 + ln];
      ushort8 pack;
#pragma unroll
      for (int r = 0; r < 8; ++r) pack[r] = f2bf(acc[nt][r] + bb);
      size_t base = (((size_t)bidx * Hc + h) * DHc + nt * 16 + ln) * Sc + srow + 8 * half;
      *(ushort8*)(VT + base) = pack;
    }
  }
}

// ---------------- fused flash attention ----------------
// grid: (Sc/128, B*H); block 256 (8 waves, same bh). Keys streamed in 32-chunks,
// K/V chunks staged cooperatively into LDS via async DMA, double buffered.
// LDS: K 2x[32 rows x 72 elem (144B pad)] | V 2x[64 rows x 40 elem (80B pad)] | P 8x512
#define KROWSTR 72
#define VROWSTR 40
#define KBUFE   (32 * KROWSTR)      // 2304 elems / buffer
#define VBUFE   (64 * VROWSTR)      // 2560 elems / buffer
__global__ __launch_bounds__(256) void k_attn(
    const unsigned short* __restrict__ Q, const unsigned short* __restrict__ Kb,
    const unsigned short* __restrict__ VT, unsigned short* __restrict__ ctx) {
  __shared__ unsigned short smem[2 * KBUFE + 2 * VBUFE + 8 * 512];
  unsigned short* ksh = smem;
  unsigned short* vsh = smem + 2 * KBUFE;
  unsigned short* pt  = smem + 2 * KBUFE + 2 * VBUFE;

  int t    = threadIdx.x;
  int wave = t >> 5;
  int lane = t & 31;
  int half = lane >> 4;
  int ln   = lane & 15;
  int bh   = blockIdx.y;
  int q0   = blockIdx.x * 128 + wave * 16;
  unsigned short* pl = pt + wave * 512;

  // ---- staging roles: 256 threads x 16B cover each 4KB chunk exactly
  int kRow = t >> 3, kSeg = t & 7;            // K chunk: 32 rows x 8 segs
  int vRow = t >> 2, vSeg = t & 3;            // V chunk: 64 rows x 4 segs
  const unsigned short* kgbase = Kb + (size_t)bh * Sc * DHc + kRow * DHc + kSeg * 8;
  const unsigned short* vgbase = VT + ((size_t)bh * DHc + vRow) * Sc + vSeg * 8;
  unsigned kldsBase = lds_addr(ksh + kRow * KROWSTR + kSeg * 8);
  unsigned vldsBase = lds_addr(vsh + vRow * VROWSTR + vSeg * 8);

  auto issue = [&](int chunk, int buf) {
    async_load16(kldsBase + buf * (KBUFE * 2), kgbase + (size_t)chunk * 32 * DHc);
    async_load16(vldsBase + buf * (VBUFE * 2), vgbase + chunk * 32);
  };

  // ---- per-wave Q fragments (K-dim = 64 -> two A frags)
  const unsigned short* qbase = Q + ((size_t)bh * Sc + q0 + ln) * DHc;
  v16bf qa0 = frag16(qbase + 16 * half);
  v16bf qa1 = frag16(qbase + 32 + 16 * half);

  v8f o[4] = {};
  float mrow[8], lrow[8];
#pragma unroll
  for (int r = 0; r < 8; ++r) { mrow[r] = -1e30f; lrow[r] = 0.f; }

  const float scale = 0.125f;                    // 1/sqrt(DH)
  const float LOG2E = 1.44269504088896340736f;

  issue(0, 0);                                   // prologue: chunk 0 -> buf 0
  for (int ci = 0; ci < Sc / 32; ++ci) {
    int nci = (ci + 1) & 63;                     // wrap: last prefetch is dead but harmless
    issue(nci, nci & 1);                         // prefetch next chunk into other buffer
    asm volatile("s_wait_asynccnt 0x2" ::: "memory");  // chunk ci's pair is done
    __syncthreads();                             // all waves' copies visible

    const unsigned short* kt = ksh + (ci & 1) * KBUFE;
    const unsigned short* vt = vsh + (ci & 1) * VBUFE;

    // ---- scores: two 16x16 tiles over keys [ci*32, ci*32+32)
    v8f s[2];
#pragma unroll
    for (int jt = 0; jt < 2; ++jt) {
      const unsigned short* kp = kt + (jt * 16 + ln) * KROWSTR;
      v8f c = {};
      c = wmma_bf16(qa0, frag16(kp + 16 * half), c);
      c = wmma_bf16(qa1, frag16(kp + 32 + 16 * half), c);
      s[jt] = c;
    }
    // ---- online softmax (row stats live per half-wave: row = r + 8*half)
    float corr[8];
#pragma unroll
    for (int r = 0; r < 8; ++r) {
      float v0 = s[0][r] * scale, v1 = s[1][r] * scale;
      float mx = fmaxf(v0, v1);
#pragma unroll
      for (int off = 1; off < 16; off <<= 1)
        mx = fmaxf(mx, __shfl_xor(mx, off, 32));
      float mnew = fmaxf(mrow[r], mx);
      float p0 = exp2f((v0 - mnew) * LOG2E);
      float p1 = exp2f((v1 - mnew) * LOG2E);
      float ps = p0 + p1;
#pragma unroll
      for (int off = 1; off < 16; off <<= 1)
        ps += __shfl_xor(ps, off, 32);
      float cr = exp2f((mrow[r] - mnew) * LOG2E);
      lrow[r] = lrow[r] * cr + ps;
      mrow[r] = mnew;
      corr[r] = cr;
      s[0][r] = p0;
      s[1][r] = p1;
    }
    // ---- repack P (C layout) -> A-fragment layout via private LDS tile
    //      (per-wave region; DS ops in a wave are in-order -> no barrier needed)
#pragma unroll
    for (int r = 0; r < 8; ++r) {
      int row = r + 8 * half;
      pl[row * 32 + ln]      = f2bf(s[0][r]);
      pl[row * 32 + 16 + ln] = f2bf(s[1][r]);
    }
    v16bf pa = frag16(pl + ln * 32 + 16 * half);
    // ---- rescale and accumulate P @ V (V from LDS)
#pragma unroll
    for (int nt = 0; nt < 4; ++nt)
#pragma unroll
      for (int r = 0; r < 8; ++r) o[nt][r] *= corr[r];
#pragma unroll
    for (int nt = 0; nt < 4; ++nt)
      o[nt] = wmma_bf16(pa, frag16(vt + (nt * 16 + ln) * VROWSTR + 16 * half), o[nt]);

    __syncthreads();   // everyone done with buf (ci&1) before it is refilled next iter
  }
  // ---- normalize, store ctx as bf16 row-major [B*S, H*DH]
  int h = bh % Hc;
  int b = bh / Hc;
#pragma unroll
  for (int nt = 0; nt < 4; ++nt) {
#pragma unroll
    for (int r = 0; r < 8; ++r) {
      float val = o[nt][r] / lrow[r];
      size_t row = (size_t)b * Sc + q0 + r + 8 * half;
      ctx[row * (Hc * DHc) + h * DHc + nt * 16 + ln] = f2bf(val);
    }
  }
}

// ---------------- output projection ----------------
// grid: (Mrows/128, 12); block 256. Wave: 16 rows x 64 cols.
__global__ __launch_bounds__(256) void k_out(
    const unsigned short* __restrict__ ctx, const unsigned short* __restrict__ WoT,
    const float* __restrict__ bo, float* __restrict__ out) {
  int wave = threadIdx.x >> 5;
  int lane = threadIdx.x & 31;
  int half = lane >> 4;
  int ln   = lane & 15;
  int m0   = blockIdx.x * 128 + wave * 16;
  int n0   = blockIdx.y * 64;

  const unsigned short* arow = ctx + (size_t)(m0 + ln) * Dc;
  v8f acc[4] = {};
  for (int kc = 0; kc < Dc; kc += 32) {
    v16bf a = frag16(arow + kc + 16 * half);
#pragma unroll
    for (int nt = 0; nt < 4; ++nt) {
      const unsigned short* bp = WoT + (size_t)(n0 + nt * 16 + ln) * Dc + kc + 16 * half;
      acc[nt] = wmma_bf16(a, frag16(bp), acc[nt]);
    }
  }
#pragma unroll
  for (int nt = 0; nt < 4; ++nt) {
    float bb = bo[n0 + nt * 16 + ln];
    size_t base = (size_t)(m0 + 8 * half) * Dc + n0 + nt * 16 + ln;
#pragma unroll
    for (int r = 0; r < 8; ++r)
      out[base + (size_t)r * Dc] = acc[nt][r] + bb;
  }
}

// ---------------- host launch ----------------
extern "C" void kernel_launch(void* const* d_in, const int* in_sizes, int n_in,
                              void* d_out, int out_size, void* d_ws, size_t ws_size,
                              hipStream_t stream) {
  const float* x  = (const float*)d_in[0];
  const float* Wq = (const float*)d_in[1];
  const float* bq = (const float*)d_in[2];
  const float* Wk = (const float*)d_in[3];
  const float* bk = (const float*)d_in[4];
  const float* Wv = (const float*)d_in[5];
  const float* bv = (const float*)d_in[6];
  const float* Wo = (const float*)d_in[7];
  const float* bo = (const float*)d_in[8];
  float* out = (float*)d_out;

  // workspace carve-up (bf16 elements); all sizes multiples of 256B
  const size_t nX  = (size_t)Mrows * Dc;        // 6291456
  const size_t nW  = (size_t)Hc * Dc * DHc;     // 589824
  const size_t nWo = (size_t)Dc * Dc;           // 589824
  const size_t nQ  = (size_t)Bc * Hc * Sc * DHc;// 6291456

  unsigned short* p = (unsigned short*)d_ws;
  unsigned short* xb   = p;            p += nX;
  unsigned short* WqT  = p;            p += nW;
  unsigned short* WkT  = p;            p += nW;
  unsigned short* WvT  = p;            p += nW;
  unsigned short* WoT  = p;            p += nWo;
  unsigned short* Qb   = p;            p += nQ;
  unsigned short* Kbb  = p;            p += nQ;
  unsigned short* VTb  = p;            p += nQ;
  unsigned short* ctxb = p;            p += nX;

  // 1) conversions
  k_cvt<<<(int)((nX + 255) / 256), 256, 0, stream>>>(x, xb, (int)nX);
  k_cvt_wt<<<(int)((nW + 255) / 256), 256, 0, stream>>>(Wq, WqT, Hc, Dc, DHc);
  k_cvt_wt<<<(int)((nW + 255) / 256), 256, 0, stream>>>(Wk, WkT, Hc, Dc, DHc);
  k_cvt_wt<<<(int)((nW + 255) / 256), 256, 0, stream>>>(Wv, WvT, Hc, Dc, DHc);
  k_cvt_wt<<<(int)((nWo + 255) / 256), 256, 0, stream>>>(Wo, WoT, 1, Dc, Dc);

  // 2) QKV projections (V stored transposed)
  k_qkv<<<dim3(Mrows / 128, Hc, 3), 256, 0, stream>>>(
      xb, WqT, WkT, WvT, bq, bk, bv, Qb, Kbb, VTb);

  // 3) fused flash attention (async double-buffered K/V staging)
  k_attn<<<dim3(Sc / 128, Bc * Hc), 256, 0, stream>>>(Qb, Kbb, VTb, ctxb);

  // 4) output projection + bias
  k_out<<<dim3(Mrows / 128, Dc / 64), 256, 0, stream>>>(ctxb, WoT, bo, out);
}
